// SparsePertokenMoE_16544214024224
// MI455X (gfx1250) — matrix operand order repaired and checked by reading, not compiled, not run on hardware
//
#include <hip/hip_runtime.h>
#include <hip/hip_bf16.h>

typedef __attribute__((ext_vector_type(16))) _Float16 v16h;
typedef __attribute__((ext_vector_type(8)))  _Float16 v8h;
typedef __attribute__((ext_vector_type(8)))  float    v8f;

#define NTOK 4096
#define DD   1024
#define HH   4096
#define ER   7

// LDS tile geometry: rows of 64 halfs (128B payload) padded to 72 halfs (144B).
// Per K-step buffer: 320 rows (A:64 + B:256) = 23040 halfs = 46080 B, x2 buffers.
#define ROWH   72
#define ROWB   144
#define BUFH   23040
#define BUFB   46080

static __device__ __forceinline__ int imin(int a, int b) { return a < b ? a : b; }

static __device__ __forceinline__ v8f wmma16(v16h a, v16h b, v8f c) {
  return __builtin_amdgcn_wmma_f32_16x16x32_f16(false, a, false, b, (short)0, c,
                                                false, false);
}

// A-fragment (16x32 f16): lane holds row m=lane&15; K halves {hi*8..} and {16+hi*8..}
static __device__ __forceinline__ v16h frag_a(const _Float16* rowk, int hi) {
  v16h r;
  *(v8h*)&r       = *(const v8h*)(rowk + hi * 8);
  *((v8h*)&r + 1) = *(const v8h*)(rowk + 16 + hi * 8);
  return r;
}

// B-fragment (32x16 f16): lane holds col n=lane&15; K halves {hi*8..} and {16+hi*8..}
static __device__ __forceinline__ v16h frag_b(const _Float16* rowk, int hi) {
  v16h r;
  const _Float16* p = rowk + hi * 8;
  *(v8h*)&r       = *(const v8h*)(p);
  *((v8h*)&r + 1) = *(const v8h*)(p + 16);
  return r;
}

// Async global -> LDS copy, 16B per lane, tracked by ASYNCcnt.
static __device__ __forceinline__ void async_b128(unsigned lds_off, const void* gaddr) {
  asm volatile("global_load_async_to_lds_b128 %0, %1, off"
               :: "v"(lds_off), "v"(gaddr)
               : "memory");
}
static __device__ __forceinline__ void wait_async0() {
  asm volatile("s_wait_asynccnt 0x0" ::: "memory");
}

// ---------------------------------------------------------------- converts ---

__global__ __launch_bounds__(256) void k_cvt_x(const float* __restrict__ s,
                                               _Float16* __restrict__ d, int n) {
  int i = (blockIdx.x * 256 + threadIdx.x) * 4;
  if (i >= n) return;
  float4 v = *(const float4*)(s + i);
  d[i + 0] = (_Float16)v.x;
  d[i + 1] = (_Float16)v.y;
  d[i + 2] = (_Float16)v.z;
  d[i + 3] = (_Float16)v.w;
}

// src f32 [R][C] (per blockIdx.z matrix) -> dst f16 [C][R]
__global__ __launch_bounds__(256) void k_tcvt(const float* __restrict__ src,
                                              _Float16* __restrict__ dst, int R, int C) {
  __shared__ _Float16 tile[64 * 66];
  size_t zoff = (size_t)blockIdx.z * R * C;
  src += zoff;
  dst += zoff;
  int rB = blockIdx.y * 64, cB = blockIdx.x * 64;
  int tid = threadIdx.x;
  int r0 = tid >> 4, c4 = (tid & 15) * 4;
#pragma unroll
  for (int i = 0; i < 4; i++) {
    int r = r0 + i * 16;
    float4 v = *(const float4*)(src + (size_t)(rB + r) * C + cB + c4);
    _Float16* t = tile + r * 66 + c4;
    t[0] = (_Float16)v.x;
    t[1] = (_Float16)v.y;
    t[2] = (_Float16)v.z;
    t[3] = (_Float16)v.w;
  }
  __syncthreads();
  int c = tid & 63, g = tid >> 6;
  __align__(16) _Float16 buf[16];
#pragma unroll
  for (int j = 0; j < 16; j++) buf[j] = tile[(g * 16 + j) * 66 + c];
  v8h* d = (v8h*)(dst + (size_t)(cB + c) * R + rB + g * 16);
  d[0] = *(v8h*)buf;
  d[1] = *((v8h*)buf + 1);
}

// ------------------------------------------------------------------ router ---

__global__ __launch_bounds__(256) void k_router(const float* __restrict__ x,
                                                const float* __restrict__ rw,
                                                int* __restrict__ eidx,
                                                float* __restrict__ scale) {
  int w = threadIdx.x >> 5, lane = threadIdx.x & 31;
  int t = blockIdx.x * 8 + w;
  const float* xr = x + (size_t)t * DD;
  float lg[8] = {0.f, 0.f, 0.f, 0.f, 0.f, 0.f, 0.f, 0.f};
#pragma unroll 4
  for (int i = 0; i < 32; i++) {
    int d = i * 32 + lane;
    float xv = xr[d];
    float4 r0 = *(const float4*)(rw + d * 8);
    float4 r1 = *(const float4*)(rw + d * 8 + 4);
    lg[0] += xv * r0.x; lg[1] += xv * r0.y; lg[2] += xv * r0.z; lg[3] += xv * r0.w;
    lg[4] += xv * r1.x; lg[5] += xv * r1.y; lg[6] += xv * r1.z; lg[7] += xv * r1.w;
  }
#pragma unroll
  for (int e = 0; e < 8; e++) {
    float v = lg[e];
    v += __shfl_xor(v, 16, 32);
    v += __shfl_xor(v, 8, 32);
    v += __shfl_xor(v, 4, 32);
    v += __shfl_xor(v, 2, 32);
    v += __shfl_xor(v, 1, 32);
    lg[e] = v;
  }
  if (lane == 0) {
    float mx = lg[0];
    int mi = 0;
#pragma unroll
    for (int e = 1; e < 8; e++)
      if (lg[e] > mx) { mx = lg[e]; mi = e; }
    float s = 0.f;
#pragma unroll
    for (int e = 0; e < 8; e++) s += __expf(lg[e] - mx);
    eidx[t] = mi;
    scale[t] = 2.0f / s;  // ALPHA * top1 prob
  }
}

// ----------------------------------------------------------------- scatter ---

__global__ void k_scatter(const int* __restrict__ idx, int* __restrict__ cnt,
                          int* __restrict__ baseArr, int* __restrict__ perm, int n) {
  int lane = threadIdx.x;  // 32 threads
  int c[ER];
#pragma unroll
  for (int k = 0; k < ER; k++) c[k] = 0;
  for (int t0 = 0; t0 < n; t0 += 32) {
    int e = idx[t0 + lane];
#pragma unroll
    for (int k = 0; k < ER; k++) {
      unsigned long long m = __ballot(e == k);
      c[k] += (int)__popcll(m);
    }
  }
  int b[ER];
  b[0] = 0;
#pragma unroll
  for (int k = 1; k < ER; k++) b[k] = b[k - 1] + c[k - 1];
  if (lane < ER) { cnt[lane] = c[lane]; baseArr[lane] = b[lane]; }
  int run[ER];
#pragma unroll
  for (int k = 0; k < ER; k++) run[k] = b[k];
  unsigned long long low = (1ull << lane) - 1ull;
  for (int t0 = 0; t0 < n; t0 += 32) {
    int e = idx[t0 + lane];
#pragma unroll
    for (int k = 0; k < ER; k++) {
      unsigned long long m = __ballot(e == k);
      if (e == k) perm[run[k] + (int)__popcll(m & low)] = t0 + lane;
      run[k] += (int)__popcll(m);
    }
  }
}

// ------------------------------------------------------------------- GEMM1 ---
// hid[row,h] = up(x) * silu(gate(x)).  WG tile 64 tok x 128 cols; wave 16x64 x {U,G}.
// LDS buffer rows: A 0..63, U 64..191, G 192..319; 64 K-halfs per stage.

__global__ __launch_bounds__(256) void k_gemm1(
    const _Float16* __restrict__ xF, const _Float16* __restrict__ upT,
    const _Float16* __restrict__ gateT, const _Float16* __restrict__ supT,
    const _Float16* __restrict__ sgateT, const int* __restrict__ cnt,
    const int* __restrict__ baseArr, const int* __restrict__ perm,
    _Float16* __restrict__ hid) {
  __shared__ _Float16 sh[2 * BUFH];
  int slot = blockIdx.z;
  int count, rowBase;
  const _Float16 *bu, *bg;
  if (slot < ER) {
    count = cnt[slot];
    rowBase = baseArr[slot];
    bu = upT + (size_t)slot * HH * DD;
    bg = gateT + (size_t)slot * HH * DD;
  } else {
    count = NTOK;
    rowBase = NTOK;
    bu = supT;
    bg = sgateT;
  }
  int mBlk = blockIdx.y;
  if (mBlk * 64 >= count) return;
  int nBlk = blockIdx.x;
  int tid = threadIdx.x, lane = tid & 31, w = tid >> 5;
  int wm = w & 3, wn = w >> 2;
  int hi = lane >> 4, nl = lane & 15, ml = lane & 15;

  // ---- per-lane async copy jobs (10 per wave, 4 rows x 128B each) ----
  unsigned shBase = (unsigned)(size_t)(const void*)sh;
  int lr = lane >> 3;              // row within job (0..3)
  int lc = (lane & 7) * 16;        // byte within row (0..112)
  const char* gjob[10];
  unsigned ljob[10];
#pragma unroll
  for (int j = 0; j < 10; j++) {
    int job = w * 10 + j;
    if (job < 16) {  // A rows (tokens)
      int row = job * 4 + lr;
      int rc = imin(mBlk * 64 + row, count - 1);
      int tok = (slot < ER) ? perm[rowBase + rc] : rc;
      gjob[j] = (const char*)(xF + (size_t)tok * DD) + lc;
      ljob[j] = shBase + (unsigned)row * ROWB + lc;
    } else if (job < 48) {  // U rows (output cols)
      int row = (job - 16) * 4 + lr;
      gjob[j] = (const char*)(bu + (size_t)(nBlk * 128 + row) * DD) + lc;
      ljob[j] = shBase + (unsigned)(64 + row) * ROWB + lc;
    } else {  // G rows
      int row = (job - 48) * 4 + lr;
      gjob[j] = (const char*)(bg + (size_t)(nBlk * 128 + row) * DD) + lc;
      ljob[j] = shBase + (unsigned)(192 + row) * ROWB + lc;
    }
  }

  v8f accU[4], accG[4];
  v8f z = {0.f, 0.f, 0.f, 0.f, 0.f, 0.f, 0.f, 0.f};
#pragma unroll
  for (int i = 0; i < 4; i++) { accU[i] = z; accG[i] = z; }

  // prologue: stage K-step 0 into buffer 0
#pragma unroll
  for (int j = 0; j < 10; j++) async_b128(ljob[j], gjob[j]);
  wait_async0();
  __syncthreads();

  const int steps = DD / 64;
  for (int i = 0; i < steps; i++) {
    int cur = i & 1;
    if (i + 1 < steps) {
      unsigned lofs = (unsigned)((i + 1) & 1) * BUFB;
      size_t gofs = (size_t)(i + 1) * 128;
#pragma unroll
      for (int j = 0; j < 10; j++) async_b128(ljob[j] + lofs, gjob[j] + gofs);
    }
    const _Float16* A0 = sh + cur * BUFH;
    const _Float16* U0 = A0 + 64 * ROWH;
    const _Float16* G0 = A0 + 192 * ROWH;
#pragma unroll
    for (int kc = 0; kc < 2; kc++) {  // two 32-K slices per stage
      v16h a = frag_a(A0 + (wm * 16 + ml) * ROWH + kc * 32, hi);
      v16h fu[4], fg[4];
#pragma unroll
      for (int nt = 0; nt < 4; nt++) {
        fu[nt] = frag_b(U0 + (wn * 64 + nt * 16 + nl) * ROWH + kc * 32, hi);
        fg[nt] = frag_b(G0 + (wn * 64 + nt * 16 + nl) * ROWH + kc * 32, hi);
      }
#pragma unroll
      for (int nt = 0; nt < 4; nt++) {
        accU[nt] = wmma16(a, fu[nt], accU[nt]);
        accG[nt] = wmma16(a, fg[nt], accG[nt]);
      }
    }
    wait_async0();
    __syncthreads();
  }

  int mTop = count - (mBlk * 64 + wm * 16);
#pragma unroll
  for (int nt = 0; nt < 4; nt++) {
#pragma unroll
    for (int r = 0; r < 8; r++) {
      int m = r + 8 * hi;
      if (m < mTop) {
        float u = accU[nt][r], g = accG[nt][r];
        float h = u * (g / (1.0f + __expf(-g)));
        int row = rowBase + mBlk * 64 + wm * 16 + m;
        int col = nBlk * 128 + wn * 64 + nt * 16 + nl;
        hid[(size_t)row * HH + col] = (_Float16)h;
      }
    }
  }
}

// ------------------------------------------------------------------- GEMM2 ---
// SHARED=true : out[t,d]  = hid_shared . sdownT   (writes every token row)
// SHARED=false: out[t,d] += scale[t] * hid_e . downT_e
// WG tile 64 tok x 256 cols; wave 16x128. LDS rows: A 0..63, B 64..319.

template <bool SHARED>
__global__ __launch_bounds__(256) void k_gemm2(
    const _Float16* __restrict__ hid, const _Float16* __restrict__ downT,
    const _Float16* __restrict__ sdownT, const int* __restrict__ cnt,
    const int* __restrict__ baseArr, const int* __restrict__ perm,
    const float* __restrict__ scale, float* __restrict__ out) {
  __shared__ _Float16 sh[2 * BUFH];
  int slot = blockIdx.z;
  int count, rowBase;
  const _Float16* bt;
  if (SHARED) {
    count = NTOK;
    rowBase = NTOK;
    bt = sdownT;
  } else {
    count = cnt[slot];
    rowBase = baseArr[slot];
    bt = downT + (size_t)slot * DD * HH;
  }
  int mBlk = blockIdx.y;
  if (mBlk * 64 >= count) return;
  int nBlk = blockIdx.x;
  int tid = threadIdx.x, lane = tid & 31, w = tid >> 5;
  int wm = w & 3, wn = w >> 2;
  int hi = lane >> 4, nl = lane & 15, ml = lane & 15;

  unsigned shBase = (unsigned)(size_t)(const void*)sh;
  int lr = lane >> 3;
  int lc = (lane & 7) * 16;
  const char* gjob[10];
  unsigned ljob[10];
#pragma unroll
  for (int j = 0; j < 10; j++) {
    int job = w * 10 + j;
    if (job < 16) {  // A rows = hid rows (already grouped per expert)
      int row = job * 4 + lr;
      int rc = imin(mBlk * 64 + row, count - 1);
      gjob[j] = (const char*)(hid + (size_t)(rowBase + rc) * HH) + lc;
      ljob[j] = shBase + (unsigned)row * ROWB + lc;
    } else {  // B rows (output cols)
      int row = (job - 16) * 4 + lr;
      gjob[j] = (const char*)(bt + (size_t)(nBlk * 256 + row) * HH) + lc;
      ljob[j] = shBase + (unsigned)(64 + row) * ROWB + lc;
    }
  }

  v8f acc[8];
  v8f z = {0.f, 0.f, 0.f, 0.f, 0.f, 0.f, 0.f, 0.f};
#pragma unroll
  for (int i = 0; i < 8; i++) acc[i] = z;

#pragma unroll
  for (int j = 0; j < 10; j++) async_b128(ljob[j], gjob[j]);
  wait_async0();
  __syncthreads();

  const int steps = HH / 64;
  for (int i = 0; i < steps; i++) {
    int cur = i & 1;
    if (i + 1 < steps) {
      unsigned lofs = (unsigned)((i + 1) & 1) * BUFB;
      size_t gofs = (size_t)(i + 1) * 128;
#pragma unroll
      for (int j = 0; j < 10; j++) async_b128(ljob[j] + lofs, gjob[j] + gofs);
    }
    const _Float16* A0 = sh + cur * BUFH;
    const _Float16* B0 = A0 + 64 * ROWH;
#pragma unroll
    for (int kc = 0; kc < 2; kc++) {
      v16h a = frag_a(A0 + (wm * 16 + ml) * ROWH + kc * 32, hi);
      v16h fb[8];
#pragma unroll
      for (int nt = 0; nt < 8; nt++)
        fb[nt] = frag_b(B0 + (wn * 128 + nt * 16 + nl) * ROWH + kc * 32, hi);
#pragma unroll
      for (int nt = 0; nt < 8; nt++) acc[nt] = wmma16(a, fb[nt], acc[nt]);
    }
    wait_async0();
    __syncthreads();
  }

  int mTop = count - (mBlk * 64 + wm * 16);
  int tok[8];
  float sc[8];
#pragma unroll
  for (int r = 0; r < 8; r++) {
    int m = r + 8 * hi;
    int mrow = mBlk * 64 + wm * 16 + m;
    int mc = imin(mrow, count - 1);
    if (SHARED) {
      tok[r] = mc;
      sc[r] = 1.f;
    } else {
      tok[r] = perm[rowBase + mc];
      sc[r] = scale[tok[r]];
    }
  }
#pragma unroll
  for (int nt = 0; nt < 8; nt++) {
#pragma unroll
    for (int r = 0; r < 8; r++) {
      int m = r + 8 * hi;
      if (m < mTop) {
        int col = nBlk * 256 + wn * 128 + nt * 16 + nl;
        size_t o = (size_t)tok[r] * DD + col;
        if (SHARED)
          out[o] = acc[nt][r];
        else
          out[o] += sc[r] * acc[nt][r];
      }
    }
  }
}

// ------------------------------------------------------------------- launch ---

extern "C" void kernel_launch(void* const* d_in, const int* in_sizes, int n_in,
                              void* d_out, int out_size, void* d_ws, size_t ws_size,
                              hipStream_t stream) {
  const float* x   = (const float*)d_in[0];
  const float* rw  = (const float*)d_in[1];
  const float* up  = (const float*)d_in[2];
  const float* gw  = (const float*)d_in[3];
  const float* dw  = (const float*)d_in[4];
  const float* sup = (const float*)d_in[5];
  const float* sg  = (const float*)d_in[6];
  const float* sd  = (const float*)d_in[7];
  float* out = (float*)d_out;

  char* ws = (char*)d_ws;
  size_t off = 0;
  auto carve = [&](size_t bytes) -> void* {
    void* p = ws + off;
    off += (bytes + 255) & ~(size_t)255;
    return p;
  };
  _Float16* xF     = (_Float16*)carve((size_t)NTOK * DD * 2);
  _Float16* upT    = (_Float16*)carve((size_t)ER * HH * DD * 2);
  _Float16* gateT  = (_Float16*)carve((size_t)ER * HH * DD * 2);
  _Float16* downT  = (_Float16*)carve((size_t)ER * DD * HH * 2);
  _Float16* supT   = (_Float16*)carve((size_t)HH * DD * 2);
  _Float16* sgateT = (_Float16*)carve((size_t)HH * DD * 2);
  _Float16* sdownT = (_Float16*)carve((size_t)DD * HH * 2);
  _Float16* hid    = (_Float16*)carve((size_t)2 * NTOK * HH * 2);
  float* scale     = (float*)carve(NTOK * 4);
  int* eidx        = (int*)carve(NTOK * 4);
  int* cnt         = (int*)carve(8 * 4);
  int* baseArr     = (int*)carve(8 * 4);
  int* perm        = (int*)carve(NTOK * 4);

  k_cvt_x<<<(NTOK * DD) / 1024, 256, 0, stream>>>(x, xF, NTOK * DD);
  k_tcvt<<<dim3(HH / 64, DD / 64, ER), 256, 0, stream>>>(up, upT, DD, HH);
  k_tcvt<<<dim3(HH / 64, DD / 64, ER), 256, 0, stream>>>(gw, gateT, DD, HH);
  k_tcvt<<<dim3(DD / 64, HH / 64, ER), 256, 0, stream>>>(dw, downT, HH, DD);
  k_tcvt<<<dim3(HH / 64, DD / 64, 1), 256, 0, stream>>>(sup, supT, DD, HH);
  k_tcvt<<<dim3(HH / 64, DD / 64, 1), 256, 0, stream>>>(sg, sgateT, DD, HH);
  k_tcvt<<<dim3(DD / 64, HH / 64, 1), 256, 0, stream>>>(sd, sdownT, HH, DD);

  k_router<<<NTOK / 8, 256, 0, stream>>>(x, rw, eidx, scale);
  k_scatter<<<1, 32, 0, stream>>>(eidx, cnt, baseArr, perm, NTOK);

  k_gemm1<<<dim3(HH / 128, NTOK / 64, 8), 256, 0, stream>>>(
      xF, upT, gateT, supT, sgateT, cnt, baseArr, perm, hid);

  k_gemm2<true><<<dim3(DD / 256, NTOK / 64, 1), 256, 0, stream>>>(
      hid, downT, sdownT, cnt, baseArr, perm, scale, out);
  k_gemm2<false><<<dim3(DD / 256, NTOK / 64, ER), 256, 0, stream>>>(
      hid, downT, sdownT, cnt, baseArr, perm, scale, out);
}
